// LSTM_Encoder_55155970015753
// MI455X (gfx1250) — matrix operand order repaired and checked by reading, not compiled, not run on hardware
//
#include <hip/hip_runtime.h>
#include <hip/hip_bf16.h>

typedef __attribute__((ext_vector_type(16))) _Float16 v16h;
typedef __attribute__((ext_vector_type(8)))  float    v8f;
typedef __attribute__((ext_vector_type(4)))  float    v4f;
typedef __attribute__((ext_vector_type(4)))  int      v4i;

#define B_  32
#define L_  128
#define H_  200
#define G4  800            // 4*H
#define M_TOT 4096         // L_*B_
#define KP0 576            // 550 padded to mult of 32
#define KP1 416            // 400 padded
#define KPH 224            // 200 padded
#define NPAIR 8001         // (L-1)(L-2)/2 for triu_indices(127,1)

#if __has_builtin(__builtin_amdgcn_global_load_async_to_lds_b128)
#define HAVE_ASYNC_LDS 1
#else
#define HAVE_ASYNC_LDS 0
#endif

#if HAVE_ASYNC_LDS
typedef __attribute__((address_space(1))) v4i* gv4i_p;
typedef __attribute__((address_space(3))) v4i* lv4i_p;
#endif

__device__ __forceinline__ float sigmoidf_(float x) {
    return 1.0f / (1.0f + __expf(-x));
}

__device__ __forceinline__ void wait_asynccnt0() {
#if HAVE_ASYNC_LDS
#if __has_builtin(__builtin_amdgcn_s_wait_asynccnt)
    __builtin_amdgcn_s_wait_asynccnt(0);
#else
    asm volatile("s_wait_asynccnt 0x0" ::: "memory");
#endif
#endif
}

// One 16x16 f32 tile of A(16xK) * B(KxN=16) accumulated over padded K.
// A row-major with stride lda; W row-major (N rows x K cols) so B[k][n] = W[n][k].
__device__ __forceinline__ v8f wmma_k_acc(const _Float16* __restrict__ Arow, int lda,
                                          const _Float16* __restrict__ Wrow, int ldw,
                                          int Kpad, v8f acc) {
    const int lane = threadIdx.x & 31;
    const int l15  = lane & 15;
    const int kh   = lane >> 4;
    const _Float16* a_base = Arow + (size_t)l15 * lda;
    const _Float16* b_base = Wrow + (size_t)l15 * ldw + kh * 16;
    for (int kb = 0; kb < Kpad; kb += 32) {
        if (kb + 32 < Kpad) __builtin_prefetch((const void*)(b_base + kb + 32), 0, 1);
        v16h a, b;
#pragma unroll
        for (int i = 0; i < 16; ++i) {
            // A 16-bit fragment: lanes 0-15 K{0..15}, lanes 16-31 K{16..31}
            int ka = kb + kh * 16 + i;
            a[i] = a_base[ka];
        }
#pragma unroll
        for (int i = 0; i < 16; ++i) b[i] = b_base[kb + i];
        acc = __builtin_amdgcn_wmma_f32_16x16x32_f16(false, a, false, b,
                                                     (short)0, acc, false, false);
    }
    return acc;
}

// ---------------- embeddings -> f16 padded X0 (M_TOT x KP0) ----------------
__global__ void embed_kernel(const int* __restrict__ lang, const int* __restrict__ word,
                             const int* __restrict__ pos,  const int* __restrict__ dep,
                             const int* __restrict__ ent,  const int* __restrict__ iob,
                             const float* __restrict__ El, const float* __restrict__ Ew,
                             const float* __restrict__ Ep, const float* __restrict__ Ed,
                             const float* __restrict__ Ee, const float* __restrict__ Ei,
                             _Float16* __restrict__ x0) {
    size_t tid = (size_t)blockIdx.x * blockDim.x + threadIdx.x;
    if (tid >= (size_t)M_TOT * KP0) return;
    int d = (int)(tid % KP0);
    int r = (int)(tid / KP0);          // r = l*32 + b
    int b = r & 31, l = r >> 5;
    float v = 0.0f;
    if (d < 550) {
        int ti = b * L_ + l;
        if (d < 50)       v = El[lang[ti] * 50  + d];
        else if (d < 350) v = Ew[(size_t)word[ti] * 300 + (d - 50)];
        else if (d < 400) v = Ep[pos[ti]  * 50  + (d - 350)];
        else if (d < 450) v = Ed[dep[ti]  * 50  + (d - 400)];
        else if (d < 500) v = Ee[ent[ti]  * 50  + (d - 450)];
        else              v = Ei[iob[ti]  * 50  + (d - 500)];
    }
    x0[tid] = (_Float16)v;
}

// ---------------- f32 -> f16 with zero K-padding (also pure zero if K==0) ----
__global__ void pad_cvt_kernel(const float* __restrict__ src, _Float16* __restrict__ dst,
                               int rows, int K, int Kpad) {
    size_t tid = (size_t)blockIdx.x * blockDim.x + threadIdx.x;
    if (tid >= (size_t)rows * Kpad) return;
    int k = (int)(tid % Kpad);
    int r = (int)(tid / Kpad);
    dst[tid] = (k < K) ? (_Float16)src[(size_t)r * K + k] : (_Float16)0.0f;
}

// ---------------- xg = X * W^T + (bih+bhh), per direction ----------------
// grid: (M_TOT/16, ceil(50/8), 2 dirs)  block: 256 (8 waves, one 16x16 tile each)
__global__ __launch_bounds__(256) void gemm_xg_kernel(
        const _Float16* __restrict__ X, const _Float16* __restrict__ W,
        const float* __restrict__ bih, const float* __restrict__ bhh,
        float* __restrict__ xg, int Kpad) {
    const int dir  = blockIdx.z;
    const int wave = threadIdx.x >> 5;
    const int lane = threadIdx.x & 31;
    const int mt   = blockIdx.x * 16;
    const int nt16 = blockIdx.y * 8 + wave;
    if (nt16 >= G4 / 16) return;            // wave-uniform: EXEC stays all-ones for WMMA
    const int nt = nt16 * 16;
    v8f acc = {};
    acc = wmma_k_acc(X + (size_t)mt * Kpad, Kpad,
                     W + ((size_t)dir * G4 + nt) * Kpad, Kpad, Kpad, acc);
    const int l15 = lane & 15, kh = lane >> 4;
    const int n = nt + l15;
    const float bias = bih[dir * G4 + n] + bhh[dir * G4 + n];
#pragma unroll
    for (int r = 0; r < 8; ++r) {
        int m = mt + r + kh * 8;
        xg[((size_t)dir * M_TOT + m) * G4 + n] = acc[r] + bias;
    }
}

// ---------------- persistent LSTM scan: one workgroup per direction ----------
// block = 1024 (32 waves). h/c live in LDS; Whh (f16, L2-resident) streamed per
// step; next step's xg slice staged into LDS via GLOBAL_LOAD_ASYNC_TO_LDS
// (overlapped with the elementwise phase).
__global__ __launch_bounds__(1024) void lstm_scan_kernel(
        const float* __restrict__ xg,        // (2, M_TOT, 800)
        const _Float16* __restrict__ Whh,    // (2, 800, 224) f16 padded
        float* __restrict__ gscr,            // (2, 32, 800) scratch
        _Float16* __restrict__ x1,           // layer0: (M_TOT, 416) f16 out
        float* __restrict__ f_out,           // layer1 dir0: (B, L, 200)
        float* __restrict__ bw_out,          // layer1 dir1: (B, L, 200)
        int layer) {
    __shared__ _Float16 hsh[B_][KPH];        // 14.3 KB
    __shared__ float    csh[B_][H_];         // 25.6 KB
#if HAVE_ASYNC_LDS
    __shared__ float    xgbuf[B_ * G4];      // 100 KB staging for one time step
#endif
    const int dir  = blockIdx.x;
    const int tid  = threadIdx.x;
    const int wave = tid >> 5;
    const int lane = tid & 31;

    for (int i = tid; i < B_ * KPH; i += 1024) hsh[i / KPH][i % KPH] = (_Float16)0.0f;
    for (int i = tid; i < B_ * H_;  i += 1024) csh[i / H_][i % H_] = 0.0f;

    const float*    xgd = xg   + (size_t)dir * M_TOT * G4;
    const _Float16* Wd  = Whh  + (size_t)dir * G4 * KPH;
    float*          gd  = gscr + (size_t)dir * B_ * G4;

#if HAVE_ASYNC_LDS
    // async-stage the first time step's gate slice into LDS
    {
        const int te0 = dir ? (L_ - 1) : 0;
        const float* src = xgd + (size_t)te0 * B_ * G4;
        for (int i = tid; i < (B_ * G4) / 4; i += 1024) {
            __builtin_amdgcn_global_load_async_to_lds_b128(
                (gv4i_p)(float*)(src + i * 4),
                (lv4i_p)(&xgbuf[i * 4]), 0, 0);
        }
        wait_asynccnt0();
    }
#endif
    __syncthreads();

    for (int t = 0; t < L_; ++t) {
        const int te = dir ? (L_ - 1 - t) : t;
        // gate matvec: 2 M-tiles x 50 N-tiles = 100 tiles over 32 waves
        for (int tile = wave; tile < 100; tile += 32) {
            const int mt = (tile & 1) * 16;
            const int nt = (tile >> 1) * 16;
            v8f acc = {};
            acc = wmma_k_acc(&hsh[mt][0], KPH, Wd + (size_t)nt * KPH, KPH, KPH, acc);
            const int l15 = lane & 15, kh = lane >> 4;
            const int n = nt + l15;
#pragma unroll
            for (int r = 0; r < 8; ++r) {
                const int m = mt + r + kh * 8;
#if HAVE_ASYNC_LDS
                gd[m * G4 + n] = acc[r] + xgbuf[m * G4 + n];
#else
                gd[m * G4 + n] = acc[r] + xgd[((size_t)te * B_ + m) * G4 + n];
#endif
            }
        }
        __threadfence();
        __syncthreads();
#if HAVE_ASYNC_LDS
        // prefetch next step's xg slice, overlapped with the elementwise phase
        if (t + 1 < L_) {
            const int ten = dir ? (L_ - 2 - t) : (t + 1);
            const float* src = xgd + (size_t)ten * B_ * G4;
            for (int i = tid; i < (B_ * G4) / 4; i += 1024) {
                __builtin_amdgcn_global_load_async_to_lds_b128(
                    (gv4i_p)(float*)(src + i * 4),
                    (lv4i_p)(&xgbuf[i * 4]), 0, 0);
            }
        }
#endif
        // elementwise cell: 32*200 = 6400 states
        for (int i = tid; i < B_ * H_; i += 1024) {
            const int m = i / H_, j = i % H_;
            const float gi = gd[m * G4 + j];
            const float gf = gd[m * G4 + H_ + j];
            const float gg = gd[m * G4 + 2 * H_ + j];
            const float go = gd[m * G4 + 3 * H_ + j];
            const float c  = sigmoidf_(gf) * csh[m][j] + sigmoidf_(gi) * tanhf(gg);
            const float h  = sigmoidf_(go) * tanhf(c);
            csh[m][j] = c;
            hsh[m][j] = (_Float16)h;
            if (layer == 0) {
                x1[((size_t)te * B_ + m) * KP1 + dir * H_ + j] = (_Float16)h;
            } else {
                float* o = dir ? bw_out : f_out;
                o[((size_t)m * L_ + te) * H_ + j] = h;
            }
        }
        wait_asynccnt0();
        __threadfence();
        __syncthreads();
    }
}

// ---------------- span features: pure streaming (410 MB write) ----------------
// f/bw are 3.3 MB each (L2-resident); output stream is written with NT hints so
// the write-once 410 MB stream does not evict them.
__global__ void span_kernel(const float* __restrict__ f, const float* __restrict__ bw,
                            float* __restrict__ out) {
    size_t tid = (size_t)blockIdx.x * blockDim.x + threadIdx.x;
    const size_t total = (size_t)NPAIR * B_ * 100;   // 100 v4f chunks per (pair,batch)
    if (tid >= total) return;
    const int c4 = (int)(tid % 100);
    size_t r = tid / 100;
    const int b = (int)(r % B_);
    const int p = (int)(r / B_);
    // invert triangular index: offset(a) = a*(253-a)/2
    int a = (int)((253.0f - sqrtf(64009.0f - 8.0f * (float)p)) * 0.5f);
    if (a < 0) a = 0; if (a > 125) a = 125;
    while (a > 0 && a * (253 - a) / 2 > p) --a;
    while (a < 125 && (a + 1) * (253 - (a + 1)) / 2 <= p) ++a;
    const int bc = a + 1 + (p - a * (253 - a) / 2);
    v4f res;
    if (c4 < 50) {
        const int ch = c4 * 4;
        const v4f* F = (const v4f*)f;
        v4f v1 = F[((size_t)(b * L_ + bc) * H_ + ch) >> 2];
        v4f v0 = F[((size_t)(b * L_ + a)  * H_ + ch) >> 2];
        res = v1 - v0;
    } else {
        const int ch = (c4 - 50) * 4;
        const v4f* Bw = (const v4f*)bw;
        v4f v1 = Bw[((size_t)(b * L_ + (a + 1))  * H_ + ch) >> 2];
        v4f v0 = Bw[((size_t)(b * L_ + (bc + 1)) * H_ + ch) >> 2];
        res = v1 - v0;
    }
    v4f* outp = (v4f*)out + (((size_t)(b * NPAIR + p) * 400 + c4 * 4) >> 2);
    __builtin_nontemporal_store(res, outp);
}

__global__ void senlen_kernel(const int* __restrict__ word, float* __restrict__ out_tail) {
    int b = threadIdx.x;
    if (b < B_) {
        int s = 0;
        for (int l = 0; l < L_; ++l) s += (word[b * L_ + l] != 0) ? 1 : 0;
        out_tail[b] = (float)(s - 2);
    }
}

extern "C" void kernel_launch(void* const* d_in, const int* in_sizes, int n_in,
                              void* d_out, int out_size, void* d_ws, size_t ws_size,
                              hipStream_t stream) {
    const int*   lang = (const int*)d_in[0];
    const int*   word = (const int*)d_in[1];
    const int*   pos  = (const int*)d_in[2];
    const int*   dep  = (const int*)d_in[3];
    const int*   ent  = (const int*)d_in[4];
    const int*   iob  = (const int*)d_in[5];
    const float* El   = (const float*)d_in[6];
    const float* Ew   = (const float*)d_in[7];
    const float* Ep   = (const float*)d_in[8];
    const float* Ed   = (const float*)d_in[9];
    const float* Ee   = (const float*)d_in[10];
    const float* Ei   = (const float*)d_in[11];
    const float* Wih0 = (const float*)d_in[12];
    const float* Whh0 = (const float*)d_in[13];
    const float* bih0 = (const float*)d_in[14];
    const float* bhh0 = (const float*)d_in[15];
    const float* Wih1 = (const float*)d_in[16];
    const float* Whh1 = (const float*)d_in[17];
    const float* bih1 = (const float*)d_in[18];
    const float* bhh1 = (const float*)d_in[19];
    float* out = (float*)d_out;

    char* ws = (char*)d_ws;
    size_t off = 0;
    auto alloc = [&](size_t bytes) { void* p = ws + off; off = (off + bytes + 255) & ~(size_t)255; return p; };
    _Float16* x0f   = (_Float16*)alloc((size_t)M_TOT * KP0 * 2);
    _Float16* Wih0f = (_Float16*)alloc((size_t)2 * G4 * KP0 * 2);
    _Float16* Whh0f = (_Float16*)alloc((size_t)2 * G4 * KPH * 2);
    _Float16* Wih1f = (_Float16*)alloc((size_t)2 * G4 * KP1 * 2);
    _Float16* Whh1f = (_Float16*)alloc((size_t)2 * G4 * KPH * 2);
    _Float16* x1f   = (_Float16*)alloc((size_t)M_TOT * KP1 * 2);
    float*    xg0   = (float*)alloc((size_t)2 * M_TOT * G4 * 4);
    float*    xg1   = (float*)alloc((size_t)2 * M_TOT * G4 * 4);
    float*    gscr  = (float*)alloc((size_t)2 * B_ * G4 * 4);
    float*    f_out = (float*)alloc((size_t)B_ * L_ * H_ * 4);
    float*    bwout = (float*)alloc((size_t)B_ * L_ * H_ * 4);

    // 1. embeddings -> f16 padded X0
    {
        size_t n = (size_t)M_TOT * KP0;
        embed_kernel<<<(unsigned)((n + 255) / 256), 256, 0, stream>>>(
            lang, word, pos, dep, ent, iob, El, Ew, Ep, Ed, Ee, Ei, x0f);
    }
    // 2. weight conversions (+ zero-fill x1 padding via K=0 trick)
    {
        size_t n;
        n = (size_t)2 * G4 * KP0; pad_cvt_kernel<<<(unsigned)((n + 255) / 256), 256, 0, stream>>>(Wih0, Wih0f, 2 * G4, 550, KP0);
        n = (size_t)2 * G4 * KPH; pad_cvt_kernel<<<(unsigned)((n + 255) / 256), 256, 0, stream>>>(Whh0, Whh0f, 2 * G4, 200, KPH);
        n = (size_t)2 * G4 * KP1; pad_cvt_kernel<<<(unsigned)((n + 255) / 256), 256, 0, stream>>>(Wih1, Wih1f, 2 * G4, 400, KP1);
        n = (size_t)2 * G4 * KPH; pad_cvt_kernel<<<(unsigned)((n + 255) / 256), 256, 0, stream>>>(Whh1, Whh1f, 2 * G4, 200, KPH);
        n = (size_t)M_TOT * KP1;  pad_cvt_kernel<<<(unsigned)((n + 255) / 256), 256, 0, stream>>>(Wih1, x1f, M_TOT, 0, KP1);
    }
    // 3. layer 0 input GEMM (both dirs), then scan
    {
        dim3 grid(M_TOT / 16, (G4 / 16 + 7) / 8, 2);
        gemm_xg_kernel<<<grid, 256, 0, stream>>>(x0f, Wih0f, bih0, bhh0, xg0, KP0);
        lstm_scan_kernel<<<2, 1024, 0, stream>>>(xg0, Whh0f, gscr, x1f, nullptr, nullptr, 0);
    }
    // 4. layer 1 input GEMM, then scan
    {
        dim3 grid(M_TOT / 16, (G4 / 16 + 7) / 8, 2);
        gemm_xg_kernel<<<grid, 256, 0, stream>>>(x1f, Wih1f, bih1, bhh1, xg1, KP1);
        lstm_scan_kernel<<<2, 1024, 0, stream>>>(xg1, Whh1f, gscr, nullptr, f_out, bwout, 1);
    }
    // 5. span features (bandwidth-dominant) + sen_lens tail
    {
        size_t n = (size_t)NPAIR * B_ * 100;
        span_kernel<<<(unsigned)((n + 255) / 256), 256, 0, stream>>>(f_out, bwout, out);
        senlen_kernel<<<1, 32, 0, stream>>>(word, out + (size_t)B_ * NPAIR * 400);
    }
}
